// SingleTokenGenerator_5016521802043
// MI455X (gfx1250) — matrix operand order repaired and checked by reading, not compiled, not run on hardware
//
#include <hip/hip_runtime.h>
#include <hip/hip_bf16.h>

#define S_LEN 1024
#define B_SZ 4
#define D_DIM 1024
#define V_SZ 32000
#define M_ROWS (B_SZ * (S_LEN - 1))   // 4092
#define M_PAD 4096
#define LN_EPS_F 1e-5f
#define EPS_LS_F 0.1f
#define NSPLIT 10
#define NTILES_TOTAL (V_SZ / 64)              // 500 tiles of 64 vocab cols
#define NT_PER_SPLIT (NTILES_TOTAL / NSPLIT)  // 50

typedef __attribute__((ext_vector_type(16))) __bf16 v16bf;
typedef __attribute__((ext_vector_type(8)))  float  v8f;

union BF16Frag {
    v16bf v;
    unsigned int u[8];
    unsigned short s[16];
};

// ---------------------------------------------------------------------------
// Zero the column-sum buffer.
// ---------------------------------------------------------------------------
__global__ void __launch_bounds__(256) zero_colsum_kernel(float* __restrict__ colsum) {
    for (int k = threadIdx.x; k < D_DIM; k += 256) colsum[k] = 0.0f;
}

// ---------------------------------------------------------------------------
// colsum[k] = sum_v W[v][k]   (chunked over V, merged via atomicAdd)
// grid = (D/256, V/256), block = 256
// ---------------------------------------------------------------------------
__global__ void __launch_bounds__(256) colsum_kernel(const float* __restrict__ W,
                                                     float* __restrict__ colsum) {
    const int k  = blockIdx.x * 256 + threadIdx.x;
    const int v0 = blockIdx.y * 256;
    float s = 0.0f;
    for (int v = v0; v < v0 + 256; ++v) s += W[(size_t)v * D_DIM + k];
    atomicAdd(&colsum[k], s);
}

// ---------------------------------------------------------------------------
// One-time fp32 -> bf16 hi/lo split of W (only when workspace is big enough).
// Each thread converts 4 consecutive elements. grid = V*D/(256*4)
// ---------------------------------------------------------------------------
__global__ void __launch_bounds__(256) wprep_kernel(const float* __restrict__ W,
                                                    unsigned short* __restrict__ Whi,
                                                    unsigned short* __restrict__ Wlo) {
    const size_t i = ((size_t)blockIdx.x * 256 + threadIdx.x) * 4;
    const float4 w = *(const float4*)(W + i);
    const float f[4] = {w.x, w.y, w.z, w.w};
    unsigned hb[4], lb[4];
#pragma unroll
    for (int j = 0; j < 4; ++j) {
        hb[j] = __float_as_uint(f[j]) & 0xFFFF0000u;
        const float lf = f[j] - __uint_as_float(hb[j]);
        lb[j] = __float_as_uint(lf);
    }
    uint2 hv, lv;
    hv.x = (hb[0] >> 16) | hb[1];
    hv.y = (hb[2] >> 16) | hb[3];
    lv.x = (lb[0] >> 16) | (lb[1] & 0xFFFF0000u);
    lv.y = (lb[2] >> 16) | (lb[3] & 0xFFFF0000u);
    *(uint2*)(Whi + i) = hv;
    *(uint2*)(Wlo + i) = lv;
}

// ---------------------------------------------------------------------------
// Per-row LayerNorm, bf16 hi/lo split of h, per-row h . colsum,
// and zero-init of sumexp / target-logit accumulators.
// grid = M_PAD blocks, block = 256
// ---------------------------------------------------------------------------
__global__ void __launch_bounds__(256) ln_prep_kernel(
    const float* __restrict__ X,       // (S, B, D)
    const float* __restrict__ gamma,
    const float* __restrict__ beta,
    const float* __restrict__ colsum,
    unsigned short* __restrict__ Hhi,
    unsigned short* __restrict__ Hlo,
    float* __restrict__ sum_logits,
    float* __restrict__ sumexp,
    float* __restrict__ targlogit)
{
    const int m   = blockIdx.x;
    const int tid = threadIdx.x;
    if (tid == 0) { sumexp[m] = 0.0f; targlogit[m] = 0.0f; }

    if (m >= M_ROWS) {  // padding rows: zero A so WMMA accumulates exp(0)=1 (masked later)
        for (int k = tid; k < D_DIM; k += 256) {
            Hhi[(size_t)m * D_DIM + k] = 0;
            Hlo[(size_t)m * D_DIM + k] = 0;
        }
        if (tid == 0) sum_logits[m] = 0.0f;
        return;
    }

    const int b = m / (S_LEN - 1);
    const int s = m % (S_LEN - 1);
    const float* x = X + ((size_t)s * B_SZ + b) * D_DIM;

    __shared__ float red0[256];
    __shared__ float red1[256];

    float xv[4];
    float lsum = 0.0f, lsq = 0.0f;
#pragma unroll
    for (int i = 0; i < 4; ++i) {
        xv[i] = x[tid + 256 * i];
        lsum += xv[i];
        lsq  += xv[i] * xv[i];
    }
    red0[tid] = lsum; red1[tid] = lsq;
    __syncthreads();
    for (int off = 128; off > 0; off >>= 1) {
        if (tid < off) { red0[tid] += red0[tid + off]; red1[tid] += red1[tid + off]; }
        __syncthreads();
    }
    const float mu  = red0[0] * (1.0f / D_DIM);
    const float var = red1[0] * (1.0f / D_DIM) - mu * mu;
    const float inv = rsqrtf(var + LN_EPS_F);
    __syncthreads();

    float dot = 0.0f;
#pragma unroll
    for (int i = 0; i < 4; ++i) {
        const int k = tid + 256 * i;
        const float h = (xv[i] - mu) * inv * gamma[k] + beta[k];
        const unsigned hb = __float_as_uint(h) & 0xFFFF0000u;
        const float hf = __uint_as_float(hb);
        const float lf = h - hf;
        Hhi[(size_t)m * D_DIM + k] = (unsigned short)(hb >> 16);
        Hlo[(size_t)m * D_DIM + k] = (unsigned short)(__float_as_uint(lf) >> 16);
        dot += h * colsum[k];
    }
    red0[tid] = dot;
    __syncthreads();
    for (int off = 128; off > 0; off >>= 1) {
        if (tid < off) red0[tid] += red0[tid + off];
        __syncthreads();
    }
    if (tid == 0) sum_logits[m] = red0[0];
}

// ---------------------------------------------------------------------------
// Fused GEMM + streaming exp-sum + target-logit pick.
// grid = (M_PAD/32, NSPLIT), block = 256 (8 waves, 2(M) x 4(N) of 16x16 tiles)
// logits z(m,v) = sum_k h[m,k] * W[v,k], via 3-product bf16 split (fp32-class).
// PRECONV: B fragments come from pre-split Whi/Wlo (pure b128 loads + WMMA);
// fallback: convert W fp32 -> bf16 hi/lo on the fly.
// ---------------------------------------------------------------------------
template <bool PRECONV>
__global__ void __launch_bounds__(256) gemm_softmax_kernel(
    const unsigned short* __restrict__ Hhi,
    const unsigned short* __restrict__ Hlo,
    const float* __restrict__ W,
    const unsigned short* __restrict__ Whi,
    const unsigned short* __restrict__ Wlo,
    const int* __restrict__ tokens,     // (B, S)
    float* __restrict__ sumexp,
    float* __restrict__ targlogit)
{
    const int lane  = threadIdx.x & 31;
    const int wave  = threadIdx.x >> 5;
    const int mw    = wave >> 2;        // 0..1  (M sub-tile)
    const int nw    = wave & 3;         // 0..3  (N sub-tile)
    const int lhalf = lane >> 4;        // 0/1
    const int lm    = lane & 15;
    const int m0    = blockIdx.x * 32 + mw * 16;

    // Target token for each of the 8 rows this lane-half owns (C-layout rows).
    int tgt[8];
#pragma unroll
    for (int r = 0; r < 8; ++r) {
        const int row = m0 + r + 8 * lhalf;
        if (row < M_ROWS) {
            const int b = row / (S_LEN - 1);
            const int s = row % (S_LEN - 1);
            tgt[r] = tokens[b * S_LEN + s + 1];
        } else {
            tgt[r] = -1;
        }
    }

    // A-fragment addressing (contiguous k order, same as B):
    //  lane<16: K = {k..k+15}; lane>=16: K = {k+16..k+31}
    const unsigned short* ahp = Hhi + (size_t)(m0 + lm) * D_DIM + lhalf * 16;
    const unsigned short* alp = Hlo + (size_t)(m0 + lm) * D_DIM + lhalf * 16;

    v8f sacc = {0.f, 0.f, 0.f, 0.f, 0.f, 0.f, 0.f, 0.f};

    for (int nt = 0; nt < NT_PER_SPLIT; ++nt) {
        const int colbase = (blockIdx.y * NT_PER_SPLIT + nt) * 64 + nw * 16;
        // B-fragment addressing (32x16): lane column = lm, 16 consecutive K.
        const size_t boff = (size_t)(colbase + lm) * D_DIM + lhalf * 16;
        const float*          wp  = W   + boff;
        const unsigned short* whp = Whi + boff;
        const unsigned short* wlp = Wlo + boff;
        if (nt + 1 < NT_PER_SPLIT) {  // global_prefetch_b8 of next tile's slab
            if (PRECONV) __builtin_prefetch(whp + (size_t)64 * D_DIM, 0, 1);
            else         __builtin_prefetch(wp  + (size_t)64 * D_DIM, 0, 1);
        }

        v8f acc = {0.f, 0.f, 0.f, 0.f, 0.f, 0.f, 0.f, 0.f};

#pragma unroll 2
        for (int kk = 0; kk < D_DIM; kk += 32) {
            BF16Frag ah, al, bh, bl;
            *(uint4*)&ah.u[0] = *(const uint4*)(ahp + kk);
            *(uint4*)&ah.u[4] = *(const uint4*)(ahp + kk + 8);
            *(uint4*)&al.u[0] = *(const uint4*)(alp + kk);
            *(uint4*)&al.u[4] = *(const uint4*)(alp + kk + 8);

            if (PRECONV) {
                // 16 consecutive pre-split bf16 per matrix: two b128 loads each.
                *(uint4*)&bh.u[0] = *(const uint4*)(whp + kk);
                *(uint4*)&bh.u[4] = *(const uint4*)(whp + kk + 8);
                *(uint4*)&bl.u[0] = *(const uint4*)(wlp + kk);
                *(uint4*)&bl.u[4] = *(const uint4*)(wlp + kk + 8);
            } else {
                const float4 w0 = *(const float4*)(wp + kk);
                const float4 w1 = *(const float4*)(wp + kk + 4);
                const float4 w2 = *(const float4*)(wp + kk + 8);
                const float4 w3 = *(const float4*)(wp + kk + 12);
                const float wv[16] = {w0.x, w0.y, w0.z, w0.w, w1.x, w1.y, w1.z, w1.w,
                                      w2.x, w2.y, w2.z, w2.w, w3.x, w3.y, w3.z, w3.w};
#pragma unroll
                for (int j = 0; j < 8; ++j) {
                    const float f0 = wv[2 * j], f1 = wv[2 * j + 1];
                    const unsigned h0 = __float_as_uint(f0) & 0xFFFF0000u;
                    const unsigned h1 = __float_as_uint(f1) & 0xFFFF0000u;
                    bh.u[j] = (h0 >> 16) | h1;
                    const float l0 = f0 - __uint_as_float(h0);
                    const float l1 = f1 - __uint_as_float(h1);
                    bl.u[j] = (__float_as_uint(l0) >> 16) |
                              (__float_as_uint(l1) & 0xFFFF0000u);
                }
            }

            acc = __builtin_amdgcn_wmma_f32_16x16x32_bf16(false, ah.v, false, bh.v,
                                                          (short)0, acc, false, false);
            acc = __builtin_amdgcn_wmma_f32_16x16x32_bf16(false, al.v, false, bh.v,
                                                          (short)0, acc, false, false);
            acc = __builtin_amdgcn_wmma_f32_16x16x32_bf16(false, ah.v, false, bl.v,
                                                          (short)0, acc, false, false);
        }

        // Streaming stats: per-slot running exp-sum + target-logit pick.
#pragma unroll
        for (int r = 0; r < 8; ++r) {
            const float z = acc[r];
            sacc[r] += __expf(z);
            if (tgt[r] == colbase + lm) {
                targlogit[m0 + r + 8 * lhalf] = z;   // unique writer
            }
        }
    }

    // Reduce per-row exp-sums across the 16 lanes of each half, merge globally.
#pragma unroll
    for (int r = 0; r < 8; ++r) {
        float v = sacc[r];
#pragma unroll
        for (int off = 1; off < 16; off <<= 1) v += __shfl_xor(v, off, 32);
        if (lm == 0) atomicAdd(&sumexp[m0 + r + 8 * lhalf], v);
    }
}

// ---------------------------------------------------------------------------
// Final masked label-smoothed loss average. block = 256, grid = 1.
// ---------------------------------------------------------------------------
__global__ void __launch_bounds__(256) loss_kernel(
    const float* __restrict__ sumexp,
    const float* __restrict__ targlogit,
    const float* __restrict__ sum_logits,
    const int* __restrict__ lengths,
    float* __restrict__ out)
{
    const int tid = threadIdx.x;
    __shared__ float rsum[256];
    __shared__ float rcnt[256];
    const float eps_i = EPS_LS_F / (float)(V_SZ - 1);
    const float c1    = 1.0f - EPS_LS_F - eps_i;
    float accs = 0.0f, accn = 0.0f;
    for (int m = tid; m < M_ROWS; m += 256) {
        const int b = m / (S_LEN - 1);
        const int s = m % (S_LEN - 1);
        if (s < lengths[b] - 1) {
            const float lse  = logf(sumexp[m]);
            const float loss = c1 * (lse - targlogit[m]) +
                               eps_i * ((float)V_SZ * lse - sum_logits[m]);
            accs += loss;
            accn += 1.0f;
        }
    }
    rsum[tid] = accs; rcnt[tid] = accn;
    __syncthreads();
    for (int off = 128; off > 0; off >>= 1) {
        if (tid < off) { rsum[tid] += rsum[tid + off]; rcnt[tid] += rcnt[tid + off]; }
        __syncthreads();
    }
    if (tid == 0) out[0] = rsum[0] / rcnt[0];
}

// ---------------------------------------------------------------------------
extern "C" void kernel_launch(void* const* d_in, const int* in_sizes, int n_in,
                              void* d_out, int out_size, void* d_ws, size_t ws_size,
                              hipStream_t stream) {
    const float* X       = (const float*)d_in[0];   // tr_hidden_state (S,B,D)
    const int*   tokens  = (const int*)d_in[1];     // (B,S)
    const int*   lengths = (const int*)d_in[2];     // (B,)
    const float* W       = (const float*)d_in[3];   // emb_weight (V,D)
    const float* gamma   = (const float*)d_in[4];
    const float* beta    = (const float*)d_in[5];
    float* out = (float*)d_out;

    char* ws = (char*)d_ws;
    unsigned short* Hhi = (unsigned short*)ws;                                   // 8 MB
    unsigned short* Hlo = (unsigned short*)(ws + (size_t)M_PAD * D_DIM * 2);     // 8 MB
    float* colsum     = (float*)(ws + (size_t)M_PAD * D_DIM * 4);                // 4 KB
    float* sum_logits = colsum + D_DIM;                                          // 16 KB
    float* sumexp     = sum_logits + M_PAD;                                      // 16 KB
    float* targlogit  = sumexp + M_PAD;                                          // 16 KB

    const size_t wsplit_off = (size_t)17 << 20;                    // 17 MB
    const size_t wbytes     = (size_t)V_SZ * D_DIM * 2;            // 64 MB each
    unsigned short* Whi = (unsigned short*)(ws + wsplit_off);
    unsigned short* Wlo = (unsigned short*)(ws + wsplit_off + wbytes);
    const bool preconv = ws_size >= wsplit_off + 2 * wbytes;       // ~145 MB

    zero_colsum_kernel<<<1, 256, 0, stream>>>(colsum);
    colsum_kernel<<<dim3(D_DIM / 256, V_SZ / 256), 256, 0, stream>>>(W, colsum);
    ln_prep_kernel<<<M_PAD, 256, 0, stream>>>(X, gamma, beta, colsum,
                                              Hhi, Hlo, sum_logits, sumexp, targlogit);
    if (preconv) {
        wprep_kernel<<<(int)(((size_t)V_SZ * D_DIM) / (256 * 4)), 256, 0, stream>>>(
            W, Whi, Wlo);
        gemm_softmax_kernel<true><<<dim3(M_PAD / 32, NSPLIT), 256, 0, stream>>>(
            Hhi, Hlo, W, Whi, Wlo, tokens, sumexp, targlogit);
    } else {
        gemm_softmax_kernel<false><<<dim3(M_PAD / 32, NSPLIT), 256, 0, stream>>>(
            Hhi, Hlo, W, Whi, Wlo, tokens, sumexp, targlogit);
    }
    loss_kernel<<<1, 256, 0, stream>>>(sumexp, targlogit, sum_logits, lengths, out);
}
